// YuanAttention_43731357008251
// MI455X (gfx1250) — matrix operand order repaired and checked by reading, not compiled, not run on hardware
//
#include <hip/hip_runtime.h>

typedef __attribute__((ext_vector_type(16))) __bf16   v16bf;
typedef __attribute__((ext_vector_type(16))) _Float16 v16h;
typedef __attribute__((ext_vector_type(8)))  float    v8f;
typedef __attribute__((ext_vector_type(4)))  int      v4i_;

constexpr int Bc = 2, Sc = 2048, Hc = 2048, NHc = 16, HDc = 128, Pc = 2048;
constexpr int Mc = Bc * Sc;                       // 4096 token rows
constexpr float SCALE = 0.08838834764831845f;     // HD^-0.5

// gfx1250 async global->LDS copy path (ASYNCcnt-tracked), with sync fallback
#if defined(__HIP_DEVICE_COMPILE__) && \
    __has_builtin(__builtin_amdgcn_global_load_async_to_lds_b128) && \
    __has_builtin(__builtin_amdgcn_s_wait_asynccnt)
#define USE_ASYNC_LDS 1
#else
#define USE_ASYNC_LDS 0
#endif

#if defined(__HIP_DEVICE_COMPILE__) && __has_builtin(__builtin_amdgcn_sched_barrier)
#define SCHED_FENCE() __builtin_amdgcn_sched_barrier(0)
#else
#define SCHED_FENCE() ((void)0)
#endif

#if USE_ASYNC_LDS
typedef __attribute__((address_space(1))) v4i_ gv4i;   // AS1: prints as 'int __vector(4) __device__'
typedef __attribute__((address_space(3))) v4i_ lv4i;   // AS3: prints as 'int __vector(4) __shared__'
__device__ __forceinline__ void async_cp16(const __bf16* g, __bf16* l) {
  __builtin_amdgcn_global_load_async_to_lds_b128(
      (gv4i*)(v4i_*)g, (lv4i*)(v4i_*)l, 0, 0);
}
#endif

// ---------------- elementwise / packing ----------------
__global__ void k_cvt_bf16(const float* __restrict__ s, __bf16* __restrict__ d, int n) {
  int i = blockIdx.x * 256 + threadIdx.x;
  if (i < n) d[i] = (__bf16)s[i];
}

// conv weight (O, K1, 2) -> (O, 2*K1) bf16, [0:K1]=tap0, [K1:2K1]=tap1
__global__ void k_pack_convw(const float* __restrict__ w, __bf16* __restrict__ d, int O, int K1) {
  int i = blockIdx.x * 256 + threadIdx.x;
  int n = O * 2 * K1;
  if (i >= n) return;
  int o = i / (2 * K1), k = i % (2 * K1);
  int tap = (k >= K1) ? 1 : 0;
  int h = tap ? (k - K1) : k;
  d[i] = (__bf16)w[((size_t)o * K1 + h) * 2 + tap];
}

// build [prev | cur] along K: prev row = s>0 ? cur[s-1] : cache[b]
__global__ void k_pack_shift(const float* __restrict__ cur, const float* __restrict__ cache,
                             __bf16* __restrict__ d, int K1) {
  int i = blockIdx.x * 256 + threadIdx.x;
  int n = Mc * 2 * K1;
  if (i >= n) return;
  int m = i / (2 * K1), k = i % (2 * K1);
  int b = m / Sc, s = m % Sc;
  float v;
  if (k < K1) v = (s > 0) ? cur[(size_t)(m - 1) * K1 + k] : cache[(size_t)b * K1 + k];
  else        v = cur[(size_t)m * K1 + (k - K1)];
  d[i] = (__bf16)v;
}

// ---------------- WMMA bf16 GEMM: C[M,N] = A[M,K] * W[N,K]^T (+bias) ----------------
// block = 8 waves, 128x128 C tile. 64-wide W k-tile staged in LDS (shared by all waves),
// double-buffered via gfx1250 async global->LDS b128 copies; A frags direct from global.
// sched_barrier pins the B-frag ds_load batch ahead of the WMMA burst so waits stagger.
__global__ __launch_bounds__(256) void k_gemm_bf16(
    const __bf16* __restrict__ A, const __bf16* __restrict__ W,
    const float* __restrict__ bias, float* __restrict__ C, int N, int K)
{
  __shared__ alignas(32) __bf16 WT[2][128][64];
  int tid = threadIdx.x;
  int lane = tid & 31, wv = tid >> 5;
  int hx = lane >> 4, lo = lane & 15;
  int m0 = blockIdx.y * 128, n0 = blockIdx.x * 128;
  const v8f vzero = {0.f,0.f,0.f,0.f,0.f,0.f,0.f,0.f};
  v8f acc[8];
#pragma unroll
  for (int t = 0; t < 8; ++t) acc[t] = vzero;
  const __bf16* arow = A + (size_t)(m0 + wv * 16 + lo) * K;
  // staging: thread covers 32 elements of one W row per 64-wide k-tile
  int wrow = tid >> 1, wseg = tid & 1;
  const __bf16* wsrc = W + (size_t)(n0 + wrow) * K + wseg * 32;
  __bf16* wd0 = &WT[0][wrow][wseg * 32];
  __bf16* wd1 = &WT[1][wrow][wseg * 32];
  int nt = K >> 6;

#if USE_ASYNC_LDS
#pragma unroll
  for (int c = 0; c < 4; ++c) async_cp16(wsrc + c * 8, wd0 + c * 8);  // fill buffer 0
#endif

  for (int it = 0; it < nt; ++it) {
    const __bf16 (*wb)[64] = WT[it & 1];
#if USE_ASYNC_LDS
    if (it + 1 < nt) {                 // kick next 64-wide tile into the other buffer
      __bf16* nd = ((it + 1) & 1) ? wd1 : wd0;
      const __bf16* ns = wsrc + (it + 1) * 64;
#pragma unroll
      for (int c = 0; c < 4; ++c) async_cp16(ns + c * 8, nd + c * 8);
      __builtin_amdgcn_s_wait_asynccnt(4);   // current tile's 4 copies complete
    } else {
      __builtin_amdgcn_s_wait_asynccnt(0);
    }
    __syncthreads();
#else
    float4 f0 = *(const float4*)(wsrc + it * 64);
    float4 f1 = *(const float4*)(wsrc + it * 64 + 8);
    float4 f2 = *(const float4*)(wsrc + it * 64 + 16);
    float4 f3 = *(const float4*)(wsrc + it * 64 + 24);
    __syncthreads();
    __bf16* nd = (it & 1) ? wd1 : wd0;
    *(float4*)nd = f0; *(float4*)(nd + 8) = f1;
    *(float4*)(nd + 16) = f2; *(float4*)(nd + 24) = f3;
    __syncthreads();
#endif
    __builtin_prefetch(arow + (it << 6) + 256, 0, 1);  // global_prefetch_b8 down A stream
#pragma unroll
    for (int kk = 0; kk < 2; ++kk) {
      int k0 = (it << 6) + (kk << 5);
      // A fragment: lane l -> row M=l&15; slots K = 16*(l>>4)+0..15
      union { v16bf v; float4 f[2]; } au;
      au.f[0] = *(const float4*)(arow + k0 + hx * 16);
      au.f[1] = *(const float4*)(arow + k0 + hx * 16 + 8);
      // B frags: lane l -> col N=l&15, K = 16*(l>>4)+j — batched loads, then WMMA burst
      v16bf bfr[8];
#pragma unroll
      for (int t = 0; t < 8; ++t)
        bfr[t] = *(const v16bf*)&wb[t * 16 + lo][(kk << 5) + hx * 16];
      SCHED_FENCE();                   // keep all ds_loads ahead of the WMMAs
#pragma unroll
      for (int t = 0; t < 8; ++t)
        acc[t] = __builtin_amdgcn_wmma_f32_16x16x32_bf16(false, au.v, false, bfr[t],
                                                         (short)0, acc[t], false, false);
      SCHED_FENCE();
    }
    __syncthreads();
  }
  int mrow0 = m0 + wv * 16 + hx * 8;  // C layout: VGPR r -> M = r + 8*(lane>>4), N = lane&15
  int ncol  = n0 + lo;
#pragma unroll
  for (int t = 0; t < 8; ++t) {
    float bb = bias ? bias[ncol + t * 16] : 0.f;
#pragma unroll
    for (int r = 0; r < 8; ++r)
      C[(size_t)(mrow0 + r) * N + ncol + t * 16] = acc[t][r] + bb;
  }
}

// ---------------- RMSNorm(y2 + x) * w -> bf16 ----------------
__global__ __launch_bounds__(256) void k_rmsnorm(const float* __restrict__ y2, const float* __restrict__ x,
                                                 const float* __restrict__ w, __bf16* __restrict__ o) {
  __shared__ float red[256];
  int m = blockIdx.x, tid = threadIdx.x;
  const float* yr = y2 + (size_t)m * Hc;
  const float* xr = x  + (size_t)m * Hc;
  float vals[8]; float ss = 0.f;
#pragma unroll
  for (int j = 0; j < 8; ++j) { float v = yr[tid + j*256] + xr[tid + j*256]; vals[j] = v; ss += v*v; }
  red[tid] = ss; __syncthreads();
  for (int k = 128; k > 0; k >>= 1) { if (tid < k) red[tid] += red[tid + k]; __syncthreads(); }
  float inv = rsqrtf(red[0] * (1.f / Hc) + 1e-6f);
#pragma unroll
  for (int j = 0; j < 8; ++j) { int c = tid + j*256; o[(size_t)m*Hc + c] = (__bf16)(vals[j] * w[c] * inv); }
}

// ---------------- RoPE: qk f32 (m, h*256+{q|k}) -> q/k f16 in (B,NH,S,HD) ----------------
__global__ void k_rope(const float* __restrict__ qk, const float* __restrict__ freqs,
                       const int* __restrict__ pos, _Float16* __restrict__ qf, _Float16* __restrict__ kf) {
  int i = blockIdx.x * 256 + threadIdx.x;
  if (i >= Mc * NHc * HDc) return;
  int d = i & (HDc - 1);
  int h = (i >> 7) & (NHc - 1);
  int m = i >> 11;
  int b = m >> 11, s = m & (Sc - 1);
  int p = pos[m];
  float f = freqs[(size_t)p * HDc + d];
  float cs = cosf(f), sn = sinf(f);
  size_t base = (size_t)m * (2 * Pc) + (size_t)h * (2 * HDc);
  float qv = qk[base + d], kv = qk[base + HDc + d];
  float qr = (d < 64) ? -qk[base + d + 64] : qk[base + d - 64];
  float kr = (d < 64) ? -qk[base + HDc + d + 64] : qk[base + HDc + d - 64];
  size_t o = ((size_t)(b * NHc + h) * Sc + s) * HDc + d;
  qf[o] = (_Float16)(qv * cs + qr * sn);
  kf[o] = (_Float16)(kv * cs + kr * sn);
}

// v f32 (m, h*128+d) -> f16 (B,NH,S,HD)
__global__ void k_v16(const float* __restrict__ v, _Float16* __restrict__ vf) {
  int i = blockIdx.x * 256 + threadIdx.x;
  if (i >= Mc * Pc) return;
  int d = i & 127, h = (i >> 7) & 15, m = i >> 11;
  int b = m >> 11, s = m & 2047;
  vf[((size_t)(b * NHc + h) * Sc + s) * HDc + d] = (_Float16)v[i];
}

// ---------------- causal flash attention, f16 WMMA, f32 online softmax ----------------
// grid: (B*NH, S/128). block: 256 = 8 waves; wave w handles q rows [q0+16w, q0+16w+16).
__global__ __launch_bounds__(256) void k_flash(const _Float16* __restrict__ Q, const _Float16* __restrict__ Km,
                                               const _Float16* __restrict__ V, __bf16* __restrict__ O) {
  __shared__ alignas(32) _Float16 VT[HDc][32];    // V transposed: VT[d][kk]
  __shared__ alignas(32) _Float16 PT[8][16][32];  // per-wave P tile for C->A relayout
  int bh = blockIdx.x, q0 = blockIdx.y * 128;
  int lane = threadIdx.x & 31, wv = threadIdx.x >> 5;
  int hx = lane >> 4, lo = lane & 15;
  const _Float16* qb = Q  + (size_t)bh * Sc * HDc;
  const _Float16* kb = Km + (size_t)bh * Sc * HDc;
  const _Float16* vb = V  + (size_t)bh * Sc * HDc;
  int qw = q0 + wv * 16;
  const v8f vzero = {0.f,0.f,0.f,0.f,0.f,0.f,0.f,0.f};
  union { v16h v; float4 f[2]; } qfr[4];
#pragma unroll
  for (int ds = 0; ds < 4; ++ds) {
    const _Float16* r = qb + (size_t)(qw + lo) * HDc + ds * 32;
    qfr[ds].f[0] = *(const float4*)(r + hx * 16);
    qfr[ds].f[1] = *(const float4*)(r + hx * 16 + 8);
  }
  v8f acc[8];
#pragma unroll
  for (int t = 0; t < 8; ++t) acc[t] = vzero;
  float mrow[8], lp[8];
#pragma unroll
  for (int r = 0; r < 8; ++r) { mrow[r] = -1e30f; lp[r] = 0.f; }
  int vrow = threadIdx.x >> 3, vseg = threadIdx.x & 7;
  int nkb = q0 / 32 + 4;                          // causal extent of this block
  for (int kbI = 0; kbI < nkb; ++kbI) {
    { // stage V tile transposed into LDS
      const _Float16* vs = vb + (size_t)(kbI * 32 + vrow) * HDc + vseg * 16;
      union { float4 f[2]; _Float16 e[16]; } u;
      u.f[0] = ((const float4*)vs)[0];
      u.f[1] = ((const float4*)vs)[1];
#pragma unroll
      for (int j = 0; j < 16; ++j) VT[vseg * 16 + j][vrow] = u.e[j];
    }
    __syncthreads();
    if (kbI * 32 <= qw + 15) {                    // wave-uniform causal skip (EXEC all-ones for WMMA)
      // batched B-frag loads (contiguous K rows), then WMMA burst
      v16h kb0[4], kb1[4];
#pragma unroll
      for (int ds = 0; ds < 4; ++ds) {
        kb0[ds] = *(const v16h*)(kb + (size_t)(kbI * 32 + lo) * HDc + ds * 32 + hx * 16);
        kb1[ds] = *(const v16h*)(kb + (size_t)(kbI * 32 + 16 + lo) * HDc + ds * 32 + hx * 16);
      }
      SCHED_FENCE();
      v8f s0 = vzero, s1 = vzero;
#pragma unroll
      for (int ds = 0; ds < 4; ++ds) {
        s0 = __builtin_amdgcn_wmma_f32_16x16x32_f16(false, qfr[ds].v, false, kb0[ds], (short)0, s0, false, false);
        s1 = __builtin_amdgcn_wmma_f32_16x16x32_f16(false, qfr[ds].v, false, kb1[ds], (short)0, s1, false, false);
      }
      SCHED_FENCE();
      int col0 = kbI * 32 + lo;
      int rb = qw + hx * 8;
      float rmax[8];
#pragma unroll
      for (int r = 0; r < 8; ++r) {
        float a = s0[r] * SCALE, c = s1[r] * SCALE;
        if (col0 > rb + r)      a = -1e30f;
        if (col0 + 16 > rb + r) c = -1e30f;
        s0[r] = a; s1[r] = c;
        rmax[r] = fmaxf(a, c);
      }
#pragma unroll
      for (int r = 0; r < 8; ++r) {               // row max across 16 columns (lanes)
        rmax[r] = fmaxf(rmax[r], __shfl_xor(rmax[r], 1, 32));
        rmax[r] = fmaxf(rmax[r], __shfl_xor(rmax[r], 2, 32));
        rmax[r] = fmaxf(rmax[r], __shfl_xor(rmax[r], 4, 32));
        rmax[r] = fmaxf(rmax[r], __shfl_xor(rmax[r], 8, 32));
      }
#pragma unroll
      for (int r = 0; r < 8; ++r) {
        float mn = fmaxf(mrow[r], rmax[r]);
        float al = __expf(mrow[r] - mn);
        mrow[r] = mn;
        float p0 = __expf(s0[r] - mn), p1 = __expf(s1[r] - mn);
        lp[r] = lp[r] * al + p0 + p1;
        s0[r] = p0; s1[r] = p1;
#pragma unroll
        for (int t = 0; t < 8; ++t) acc[t][r] *= al;
      }
      // C-layout P -> A-layout frag via per-wave LDS tile (in-order DS within a wave)
#pragma unroll
      for (int r = 0; r < 8; ++r) {
        PT[wv][hx * 8 + r][lo]      = (_Float16)s0[r];
        PT[wv][hx * 8 + r][16 + lo] = (_Float16)s1[r];
      }
      __builtin_amdgcn_wave_barrier();
      asm volatile("s_wait_dscnt 0" ::: "memory");
      union { v16h v; float4 f[2]; } pf;
      pf.f[0] = *(const float4*)&PT[wv][lo][hx * 16];
      pf.f[1] = *(const float4*)&PT[wv][lo][hx * 16 + 8];
      v16h bvr[8];                                // batched V B-frags from LDS
#pragma unroll
      for (int t = 0; t < 8; ++t)
        bvr[t] = *(const v16h*)&VT[t * 16 + lo][hx * 16];
      SCHED_FENCE();
#pragma unroll
      for (int t = 0; t < 8; ++t)
        acc[t] = __builtin_amdgcn_wmma_f32_16x16x32_f16(false, pf.v, false, bvr[t], (short)0, acc[t], false, false);
      SCHED_FENCE();
    }
    __syncthreads();
  }
#pragma unroll
  for (int r = 0; r < 8; ++r) {                   // row sum of l across 16 lanes
    lp[r] += __shfl_xor(lp[r], 1, 32);
    lp[r] += __shfl_xor(lp[r], 2, 32);
    lp[r] += __shfl_xor(lp[r], 4, 32);
    lp[r] += __shfl_xor(lp[r], 8, 32);
  }
  int b = bh / NHc, h = bh % NHc;
#pragma unroll
  for (int r = 0; r < 8; ++r) {
    int row = qw + hx * 8 + r;
    float inv = 1.f / lp[r];
    size_t ob = ((size_t)(b * Sc + row)) * Pc + (size_t)h * HDc + lo;
#pragma unroll
    for (int t = 0; t < 8; ++t) O[ob + t * 16] = (__bf16)(acc[t][r] * inv);
  }
}

// ---------------- cache tails: lf1_new = x[:,S-1], lf2_new = y1[:,S-1] ----------------
__global__ void k_tail(const float* __restrict__ hs, const float* __restrict__ y1, float* __restrict__ out) {
  int i = blockIdx.x * 256 + threadIdx.x;
  size_t base = (size_t)Mc * Hc;
  if (i < Bc * Hc) {
    int b = i / Hc, h = i % Hc;
    out[base + i] = hs[((size_t)(b * Sc + Sc - 1)) * Hc + h];
  } else if (i < Bc * Hc + Bc * (Hc / 2)) {
    int j = i - Bc * Hc;
    int b = j / (Hc / 2), o = j % (Hc / 2);
    out[base + i] = y1[((size_t)(b * Sc + Sc - 1)) * (Hc / 2) + o];
  }
}

extern "C" void kernel_launch(void* const* d_in, const int* in_sizes, int n_in,
                              void* d_out, int out_size, void* d_ws, size_t ws_size,
                              hipStream_t stream) {
  const float* hs    = (const float*)d_in[0];
  const float* freqs = (const float*)d_in[1];
  const int*   pos   = (const int*)d_in[2];
  const float* Wqk   = (const float*)d_in[3];
  const float* Wv    = (const float*)d_in[4];
  const float* Wo    = (const float*)d_in[5];
  const float* c1w   = (const float*)d_in[6];
  const float* c1b   = (const float*)d_in[7];
  const float* c2w   = (const float*)d_in[8];
  const float* c2b   = (const float*)d_in[9];
  const float* lnw   = (const float*)d_in[10];
  const float* lf1   = (const float*)d_in[11];
  const float* lf2   = (const float*)d_in[12];
  float* out = (float*)d_out;

  char* ws = (char*)d_ws; size_t off = 0;
  auto take = [&](size_t bytes) -> void* { void* p = ws + off; off += (bytes + 255) & ~(size_t)255; return p; };

  __bf16*   hs_bf  = (__bf16*)take((size_t)Mc * Hc * 2);
  __bf16*   Wv_bf  = (__bf16*)take((size_t)Pc * Hc * 2);
  __bf16*   Wqk_bf = (__bf16*)take((size_t)2 * Pc * Hc * 2);
  __bf16*   Wo_bf  = (__bf16*)take((size_t)Hc * Pc * 2);
  __bf16*   W1c    = (__bf16*)take((size_t)(Hc / 2) * (2 * Hc) * 2);
  __bf16*   W2c    = (__bf16*)take((size_t)Hc * Hc * 2);
  __bf16*   A2     = (__bf16*)take((size_t)Mc * (2 * Hc) * 2);
  __bf16*   A3     = (__bf16*)take((size_t)Mc * Hc * 2);
  float*    v32    = (float*)take((size_t)Mc * Pc * 4);
  float*    y1     = (float*)take((size_t)Mc * (Hc / 2) * 4);
  float*    y2     = (float*)take((size_t)Mc * Hc * 4);
  __bf16*   lfo    = (__bf16*)take((size_t)Mc * Hc * 2);
  float*    qk     = (float*)take((size_t)Mc * 2 * Pc * 4);
  _Float16* qf     = (_Float16*)take((size_t)Mc * Pc * 2);
  _Float16* kf     = (_Float16*)take((size_t)Mc * Pc * 2);
  _Float16* vf     = (_Float16*)take((size_t)Mc * Pc * 2);
  __bf16*   attn   = (__bf16*)take((size_t)Mc * Pc * 2);

  dim3 T(256);
  auto nb = [](long n) { return dim3((unsigned)((n + 255) / 256)); };

  // precision packing
  k_cvt_bf16<<<nb((long)Mc * Hc), T, 0, stream>>>(hs, hs_bf, Mc * Hc);
  k_cvt_bf16<<<nb((long)Pc * Hc), T, 0, stream>>>(Wv, Wv_bf, Pc * Hc);
  k_cvt_bf16<<<nb((long)2 * Pc * Hc), T, 0, stream>>>(Wqk, Wqk_bf, 2 * Pc * Hc);
  k_cvt_bf16<<<nb((long)Hc * Pc), T, 0, stream>>>(Wo, Wo_bf, Hc * Pc);
  k_pack_convw<<<nb((long)(Hc / 2) * 2 * Hc), T, 0, stream>>>(c1w, W1c, Hc / 2, Hc);
  k_pack_convw<<<nb((long)Hc * Hc), T, 0, stream>>>(c2w, W2c, Hc, Hc / 2);

  // v = hs @ Wv^T
  k_gemm_bf16<<<dim3(Pc / 128, Mc / 128), T, 0, stream>>>(hs_bf, Wv_bf, nullptr, v32, Pc, Hc);

  // conv1 as GEMM over [prev|cur] with [w1a|w1b]
  k_pack_shift<<<nb((long)Mc * 2 * Hc), T, 0, stream>>>(hs, lf1, A2, Hc);
  k_gemm_bf16<<<dim3((Hc / 2) / 128, Mc / 128), T, 0, stream>>>(A2, W1c, c1b, y1, Hc / 2, 2 * Hc);
  // conv2 as GEMM over [y1_prev|y1] with [w2a|w2b]
  k_pack_shift<<<nb((long)Mc * Hc), T, 0, stream>>>(y1, lf2, A3, Hc / 2);
  k_gemm_bf16<<<dim3(Hc / 128, Mc / 128), T, 0, stream>>>(A3, W2c, c2b, y2, Hc, Hc);

  // lf_out = rmsnorm(y2 + x); qk = lf_out @ Wqk^T
  k_rmsnorm<<<dim3(Mc), T, 0, stream>>>(y2, hs, lnw, lfo);
  k_gemm_bf16<<<dim3((2 * Pc) / 128, Mc / 128), T, 0, stream>>>(lfo, Wqk_bf, nullptr, qk, 2 * Pc, Hc);

  // RoPE + f16 packing, then causal flash attention
  k_rope<<<nb((long)Mc * NHc * HDc), T, 0, stream>>>(qk, freqs, pos, qf, kf);
  k_v16<<<nb((long)Mc * Pc), T, 0, stream>>>(v32, vf);
  k_flash<<<dim3(Bc * NHc, Sc / 128), T, 0, stream>>>(qf, kf, vf, attn);

  // output = attn @ Wo^T, plus cache tails
  k_gemm_bf16<<<dim3(Hc / 128, Mc / 128), T, 0, stream>>>(attn, Wo_bf, nullptr, out, Hc, Pc);
  k_tail<<<dim3(24), T, 0, stream>>>(hs, y1, out);
}
